// HypergraphEncoder_57234734187132
// MI455X (gfx1250) — matrix operand order repaired and checked by reading, not compiled, not run on hardware
//
#include <hip/hip_runtime.h>
#include <hip/hip_bf16.h>

typedef __attribute__((ext_vector_type(16))) __bf16 v16bf;
typedef __attribute__((ext_vector_type(8)))  float  v8f;

#define N_NODES  100000
#define N_HEDGES 25000
#define N_INC    1600000
#define IN_DIM   256
#define H_DIM    128

// ---------------------------------------------------------------------------
// Small prep: W_comb = W_n2e @ W_in  (f32), b_comb = W_n2e@b_in + b_n2e
// ---------------------------------------------------------------------------
__global__ __launch_bounds__(256)
void k_prep_wcomb(const float* __restrict__ W_in, const float* __restrict__ b_in,
                  const float* __restrict__ W_n2e, const float* __restrict__ b_n2e,
                  float* __restrict__ Wcomb, float* __restrict__ b_comb) {
  __shared__ float wrow[H_DIM];
  const int n = blockIdx.x;       // output row 0..127
  const int j = threadIdx.x;      // output col 0..255
  if (j < H_DIM) wrow[j] = W_n2e[n * H_DIM + j];
  __syncthreads();
  float acc = 0.f;
  #pragma unroll 4
  for (int k = 0; k < H_DIM; ++k) acc += wrow[k] * W_in[k * IN_DIM + j];
  Wcomb[n * IN_DIM + j] = acc;
  if (j == 0) {
    float bb = b_n2e[n];
    for (int k = 0; k < H_DIM; ++k) bb += wrow[k] * b_in[k];
    b_comb[n] = bb;
  }
}

// ---------------------------------------------------------------------------
// Pack a [H_DIM x K] f32 weight matrix into fragment-linear bf16 layout:
// fragment (kb, nt): 32 lanes x 16 bf16 contiguous.
// Element B[n][k] -> lane = (n%16) + 16*((k%32)/16), e = k%16 within lane.
// ---------------------------------------------------------------------------
__global__ __launch_bounds__(256)
void k_pack_w(const float* __restrict__ W, __bf16* __restrict__ Wpk, int K) {
  const int i = blockIdx.x * 256 + threadIdx.x;
  if (i >= H_DIM * K) return;
  const int n = i / K, k = i % K;
  const int nt = n >> 4, l16 = n & 15;
  const int kb = k >> 5, rem = k & 31;
  const int hi = rem >> 4, e = rem & 15;
  Wpk[(((size_t)kb * 8 + nt) * 32 + (l16 + hi * 16)) * 16 + e] = (__bf16)W[i];
}

__global__ __launch_bounds__(256)
void k_inv(const float* __restrict__ in, float* __restrict__ out, int n) {
  const int i = blockIdx.x * 256 + threadIdx.x;
  if (i < n) out[i] = 1.0f / in[i];
}

__global__ __launch_bounds__(256)
void k_zero(float* __restrict__ p, int n) {
  const int i = blockIdx.x * 256 + threadIdx.x;
  if (i < n) p[i] = 0.f;
}

__global__ __launch_bounds__(256)
void k_prelu(const float* __restrict__ in, float* __restrict__ out,
             const float* __restrict__ pa, int n) {
  const int i = blockIdx.x * 256 + threadIdx.x;
  if (i < n) {
    const float a = *pa;
    const float x = in[i];
    out[i] = (x >= 0.f) ? x : a * x;
  }
}

// ---------------------------------------------------------------------------
// WMMA GEMM:  C[m][n] = (sum_k A[m][k]*B[n][k] + bias[n]) * rowScale[m]
// A: [M x K] f32 row-major (converted to bf16 on the fly)
// Bpk: packed bf16 fragments of B [H_DIM x K]
// One wave computes a 16(M) x 128(N) strip; 8 waves/block -> 128 rows/block.
// All 8 B fragments of a k-step are loaded up front (one load clause) so the
// 8 dependent WMMAs can drain behind staged s_wait_loadcnt counters.
// ---------------------------------------------------------------------------
__global__ __launch_bounds__(256)
void k_gemm_bf16(const float* __restrict__ A, const __bf16* __restrict__ Bpk,
                 const float* __restrict__ bias, const float* __restrict__ rowScale,
                 float* __restrict__ C, int M, int K) {
  const int tid  = threadIdx.x;
  const int wave = tid >> 5;
  const int lane = tid & 31;
  const int l16  = lane & 15;
  const int hi   = lane >> 4;          // 0 or 1
  const int m0   = blockIdx.x * 128 + wave * 16;

  v8f acc[8] = {};

  const int row = m0 + l16;
  const float* Ar = A + (size_t)(row < M ? row : 0) * K;
  const float msk = (row < M) ? 1.f : 0.f;

  for (int kb = 0; kb < K; kb += 32) {
    // Batch-load all 8 B fragments for this k-step (independent b128 loads)
    const v16bf* bbase = ((const v16bf*)Bpk) + ((size_t)(kb >> 5) * 8) * 32 + lane;
    v16bf bfr[8];
    #pragma unroll
    for (int nt = 0; nt < 8; ++nt) bfr[nt] = bbase[nt * 32];

    // A fragment: contiguous k order matching packed B (lanes 0-15: K kb..kb+15;
    // lanes 16-31: K kb+16..kb+31)
    const int k0 = kb + hi * 16;
    const float4 a0 = *(const float4*)(Ar + k0);
    const float4 a1 = *(const float4*)(Ar + k0 + 4);
    const float4 a2 = *(const float4*)(Ar + k0 + 8);
    const float4 a3 = *(const float4*)(Ar + k0 + 12);
    v16bf af;
    af[0]  = (__bf16)(a0.x * msk); af[1]  = (__bf16)(a0.y * msk);
    af[2]  = (__bf16)(a0.z * msk); af[3]  = (__bf16)(a0.w * msk);
    af[4]  = (__bf16)(a1.x * msk); af[5]  = (__bf16)(a1.y * msk);
    af[6]  = (__bf16)(a1.z * msk); af[7]  = (__bf16)(a1.w * msk);
    af[8]  = (__bf16)(a2.x * msk); af[9]  = (__bf16)(a2.y * msk);
    af[10] = (__bf16)(a2.z * msk); af[11] = (__bf16)(a2.w * msk);
    af[12] = (__bf16)(a3.x * msk); af[13] = (__bf16)(a3.y * msk);
    af[14] = (__bf16)(a3.z * msk); af[15] = (__bf16)(a3.w * msk);

    #pragma unroll
    for (int nt = 0; nt < 8; ++nt) {
      acc[nt] = __builtin_amdgcn_wmma_f32_16x16x32_bf16(
          false, af, false, bfr[nt], (short)0, acc[nt], false, false);
    }
  }

  // epilogue: C/D layout -> VGPR r holds M = m0 + r + 8*hi, N = nt*16 + l16
  float sc[8];
  #pragma unroll
  for (int r = 0; r < 8; ++r) {
    const int m = m0 + r + hi * 8;
    sc[r] = (m < M) ? rowScale[m] : 0.f;
  }
  #pragma unroll
  for (int nt = 0; nt < 8; ++nt) {
    const int n = nt * 16 + l16;
    const float bn = bias[n];
    #pragma unroll
    for (int r = 0; r < 8; ++r) {
      const int m = m0 + r + hi * 8;
      if (m < M) C[(size_t)m * H_DIM + n] = (acc[nt][r] + bn) * sc[r];
    }
  }
}

// ---------------------------------------------------------------------------
// Scatter: one wave per incidence pair, 4 floats per lane.
// out[s][:] += invw[s] * msg[g][:]
// ---------------------------------------------------------------------------
__global__ __launch_bounds__(256)
void k_scatter(const float* __restrict__ msg, const int* __restrict__ gidx,
               const int* __restrict__ sidx, const float* __restrict__ invw,
               float* __restrict__ out, int n) {
  const int inc = blockIdx.x * 8 + (threadIdx.x >> 5);
  if (inc >= n) return;
  const int lane = threadIdx.x & 31;
  const int g = gidx[inc];
  const int s = sidx[inc];
  const float w = invw[s];
  const float4 v = *(const float4*)(msg + (size_t)g * H_DIM + lane * 4);
  float* o = out + (size_t)s * H_DIM + lane * 4;
  unsafeAtomicAdd(o + 0, w * v.x);
  unsafeAtomicAdd(o + 1, w * v.y);
  unsafeAtomicAdd(o + 2, w * v.z);
  unsafeAtomicAdd(o + 3, w * v.w);
}

// ---------------------------------------------------------------------------
extern "C" void kernel_launch(void* const* d_in, const int* in_sizes, int n_in,
                              void* d_out, int out_size, void* d_ws, size_t ws_size,
                              hipStream_t stream) {
  (void)in_sizes; (void)n_in; (void)out_size; (void)ws_size;
  const float* n_feat        = (const float*)d_in[0];
  // d_in[1] he_feat: unused by reference
  const float* node_norm     = (const float*)d_in[2];
  const float* node_norm_sum = (const float*)d_in[3];
  const float* edge_norm     = (const float*)d_in[4];
  const float* edge_norm_sum = (const float*)d_in[5];
  const float* W_in          = (const float*)d_in[6];
  const float* b_in          = (const float*)d_in[7];
  const float* W_n2e         = (const float*)d_in[8];
  const float* b_n2e         = (const float*)d_in[9];
  const float* W_e2n         = (const float*)d_in[10];
  const float* b_e2n         = (const float*)d_in[11];
  const float* prelu_a       = (const float*)d_in[12];
  const int*   src_nodes     = (const int*)d_in[13];
  const int*   dst_hedges    = (const int*)d_in[14];

  float* out       = (float*)d_out;
  float* nfeat_out = out;                              // [100000 x 128]
  float* efeat     = out + (size_t)N_NODES * H_DIM;    // [25000 x 128]

  char* ws = (char*)d_ws;
  float*  Wh        = (float*)(ws);                    // 51.2 MB  (aliased: h_n later)
  float*  h_n       = Wh;
  float*  h_e       = (float*)(ws + 51200000);         // 12.8 MB
  float*  Wh2       = (float*)(ws + 64000000);         // 12.8 MB
  char*   tail      = ws + 76800000;
  float*  Wcomb_f32 = (float*)(tail);                  // 131072 B
  __bf16* WpkA      = (__bf16*)(tail + 131072);        // 65536 B
  __bf16* WpkB      = (__bf16*)(tail + 196608);        // 32768 B
  float*  b_comb    = (float*)(tail + 229376);         // 512 B
  float*  inv_ens   = (float*)(tail + 229888);         // 100000 B
  float*  inv_nns   = (float*)(tail + 329888);         // 400000 B

  // 1. weight prep (fused W_comb = W_n2e@W_in, packed bf16 fragments, recips)
  k_prep_wcomb<<<H_DIM, 256, 0, stream>>>(W_in, b_in, W_n2e, b_n2e, Wcomb_f32, b_comb);
  k_pack_w<<<(H_DIM * IN_DIM + 255) / 256, 256, 0, stream>>>(Wcomb_f32, WpkA, IN_DIM);
  k_pack_w<<<(H_DIM * H_DIM + 255) / 256, 256, 0, stream>>>(W_e2n, WpkB, H_DIM);
  k_inv<<<(N_HEDGES + 255) / 256, 256, 0, stream>>>(edge_norm_sum, inv_ens, N_HEDGES);
  k_inv<<<(N_NODES + 255) / 256, 256, 0, stream>>>(node_norm_sum, inv_nns, N_NODES);
  k_zero<<<(N_HEDGES * H_DIM + 255) / 256, 256, 0, stream>>>(h_e, N_HEDGES * H_DIM);

  // 2. Wh_scaled = node_norm * (n_feat @ W_combT + b_comb)     [fused 2 GEMMs]
  k_gemm_bf16<<<(N_NODES + 127) / 128, 256, 0, stream>>>(
      n_feat, WpkA, b_comb, node_norm, Wh, N_NODES, IN_DIM);

  // 3. node -> hyperedge scatter + PReLU -> efeat (output #2)
  k_scatter<<<(N_INC + 7) / 8, 256, 0, stream>>>(
      Wh, src_nodes, dst_hedges, inv_ens, h_e, N_INC);
  k_prelu<<<(N_HEDGES * H_DIM + 255) / 256, 256, 0, stream>>>(
      h_e, efeat, prelu_a, N_HEDGES * H_DIM);

  // 4. Wh2_scaled = edge_norm * (efeat @ W_e2nT + b_e2n)
  k_gemm_bf16<<<(N_HEDGES + 127) / 128, 256, 0, stream>>>(
      efeat, WpkB, b_e2n, edge_norm, Wh2, N_HEDGES, H_DIM);

  // 5. hyperedge -> node scatter + PReLU -> nfeat_out (output #1)
  // (h_n aliases Wh; Wh is dead after step 3, stream order guarantees safety)
  k_zero<<<(N_NODES * H_DIM + 255) / 256, 256, 0, stream>>>(h_n, N_NODES * H_DIM);
  k_scatter<<<(N_INC + 7) / 8, 256, 0, stream>>>(
      Wh2, dst_hedges, src_nodes, inv_nns, h_n, N_INC);
  k_prelu<<<(N_NODES * H_DIM + 255) / 256, 256, 0, stream>>>(
      h_n, nfeat_out, prelu_a, N_NODES * H_DIM);
}
